// CausalMultiHeadAttention_74586402062365
// MI455X (gfx1250) — compile-verified, matrix operand order checked by reading, not run on hardware
//
#include <hip/hip_runtime.h>

// ---------------------------------------------------------------------------
// Causal MHA forward, MI455X (gfx1250, wave32, WMMA bf16)
// B=4 S=2048 E=1024 H=16 D=64
// ---------------------------------------------------------------------------

typedef __bf16 bf16_t;
typedef __attribute__((ext_vector_type(16))) __bf16 v16bf;
typedef __attribute__((ext_vector_type(8)))  __bf16 v8bf;
typedef __attribute__((ext_vector_type(8)))  float  v8f;

#define CB  4
#define CS  2048
#define CE  1024
#define CH  16
#define CD  64
#define CM  (CB * CS)          // 8192 rows

static __device__ __forceinline__ v16bf cat8(v8bf lo, v8bf hi) {
  return __builtin_shufflevector(lo, hi, 0,1,2,3,4,5,6,7,8,9,10,11,12,13,14,15);
}

static __device__ __forceinline__ v16bf load_frag(const bf16_t* p) {
  // 16 bf16 per lane as two contiguous 16B chunks (K .. K+7, K+16 .. K+23)
  v8bf lo = *(const v8bf*)p;
  v8bf hi = *(const v8bf*)(p + 16);
  return cat8(lo, hi);
}

#define WMMA_BF16(A, B, C) \
  __builtin_amdgcn_wmma_f32_16x16x32_bf16(false, (A), false, (B), (short)0, (C), false, false)

// ---------------------------------------------------------------------------
// fp32 -> bf16 conversion
// ---------------------------------------------------------------------------
__global__ __launch_bounds__(256) void cvt_bf16_kernel(const float* __restrict__ src,
                                                       bf16_t* __restrict__ dst, int n) {
  int i = blockIdx.x * blockDim.x + threadIdx.x;
  if (i < n) dst[i] = (bf16_t)src[i];
}

// ---------------------------------------------------------------------------
// QKV projection: qkv[m,f] = sum_e x[m,e]*Wa[f,e] + b[f]
// One wave -> 32(M) x 64(N) tile.  Ping-pong double-buffered k-loop, FULLY
// UNROLLED (16 steps): all load addresses are immediate offsets off
// loop-invariant bases -> no per-step address VALU, no WMMA WAR-hazard NOPs.
// Writes Q,K as [B,H,S,D], V as [B,H,D,S].
// ---------------------------------------------------------------------------
__global__ __launch_bounds__(256) void gemm_qkv_kernel(
    const bf16_t* __restrict__ xb, const bf16_t* __restrict__ Wab,
    const float* __restrict__ bias,
    bf16_t* __restrict__ Qb, bf16_t* __restrict__ Kb, bf16_t* __restrict__ Vt) {
  const int lane = threadIdx.x & 31;
  const int wave = blockIdx.x * (blockDim.x >> 5) + (threadIdx.x >> 5);
  const int NT = (3 * CE) / 64;             // 48 n-tiles
  const int m0 = (wave / NT) * 32;
  const int n0 = (wave % NT) * 64;
  const int r  = lane & 15;
  const int ks = lane >> 4;

  v8f acc[8];
  const v8f vzero = {0.f,0.f,0.f,0.f,0.f,0.f,0.f,0.f};
#pragma unroll
  for (int i = 0; i < 8; ++i) acc[i] = vzero;

  const bf16_t* arow0 = xb + (size_t)(m0 + r) * CE + ks * 8;
  const bf16_t* arow1 = xb + (size_t)(m0 + 16 + r) * CE + ks * 8;
  const bf16_t* brow[4];
#pragma unroll
  for (int nn = 0; nn < 4; ++nn)
    brow[nn] = Wab + (size_t)(n0 + nn * 16 + r) * CE + ks * 8;

  // ping buffers (k = kk)
  v16bf pa0 = load_frag(arow0), pa1 = load_frag(arow1);
  v16bf pb[4];
#pragma unroll
  for (int nn = 0; nn < 4; ++nn) pb[nn] = load_frag(brow[nn]);

#pragma unroll
  for (int kk = 0; kk < CE; kk += 64) {
    const int k1 = kk + 32;
    // pong buffers (k = kk+32)
    v16bf qa0 = load_frag(arow0 + k1);
    v16bf qa1 = load_frag(arow1 + k1);
    v16bf qb[4];
#pragma unroll
    for (int nn = 0; nn < 4; ++nn) qb[nn] = load_frag(brow[nn] + k1);
    // consume ping
#pragma unroll
    for (int nn = 0; nn < 4; ++nn) acc[nn]     = WMMA_BF16(pa0, pb[nn], acc[nn]);
#pragma unroll
    for (int nn = 0; nn < 4; ++nn) acc[4 + nn] = WMMA_BF16(pa1, pb[nn], acc[4 + nn]);
    // refill ping (k = kk+64) -- statically elided on the last step
    if (kk + 64 < CE) {
      const int k2 = kk + 64;
      pa0 = load_frag(arow0 + k2);
      pa1 = load_frag(arow1 + k2);
#pragma unroll
      for (int nn = 0; nn < 4; ++nn) pb[nn] = load_frag(brow[nn] + k2);
    }
    // consume pong
#pragma unroll
    for (int nn = 0; nn < 4; ++nn) acc[nn]     = WMMA_BF16(qa0, qb[nn], acc[nn]);
#pragma unroll
    for (int nn = 0; nn < 4; ++nn) acc[4 + nn] = WMMA_BF16(qa1, qb[nn], acc[4 + nn]);
  }

#pragma unroll
  for (int mt = 0; mt < 2; ++mt) {
#pragma unroll
    for (int nn = 0; nn < 4; ++nn) {
      const int f    = n0 + nn * 16 + r;
      const int sect = f >> 10;            // 0=Q 1=K 2=V
      const int e    = f & (CE - 1);
      const int hh   = e >> 6;
      const int d    = e & (CD - 1);
      const float bv = bias[f];
#pragma unroll
      for (int v = 0; v < 8; ++v) {
        const int m  = m0 + mt * 16 + v + 8 * ks;
        const int bb = m >> 11;            // / S
        const int s  = m & (CS - 1);
        const float val = acc[mt * 4 + nn][v] + bv;
        if (sect == 0)
          Qb[(((size_t)(bb * CH + hh) * CS) + s) * CD + d] = (bf16_t)val;
        else if (sect == 1)
          Kb[(((size_t)(bb * CH + hh) * CS) + s) * CD + d] = (bf16_t)val;
        else
          Vt[(((size_t)(bb * CH + hh) * CD) + d) * CS + s] = (bf16_t)val;
      }
    }
  }
}

// ---------------------------------------------------------------------------
// Flash attention: one wave = one (b, h, 16-query tile).
// Computes S^T = K*Q^T so the score C-fragment IS the P B-fragment for
// O^T += V^T * P (no transposes, no LDS).  Online softmax in f32.
// Pipelined: V fragments load at the top of each step (hidden behind the
// score WMMAs + softmax); next step's K fragments are register-staged and
// hidden behind softmax + PV WMMA; next V columns are prefetched.
// ---------------------------------------------------------------------------
__global__ __launch_bounds__(256) void attn_kernel(
    const bf16_t* __restrict__ Qb, const bf16_t* __restrict__ Kb,
    const bf16_t* __restrict__ Vt, bf16_t* __restrict__ Ob) {
  const int lane = threadIdx.x & 31;
  const int wave = blockIdx.x * (blockDim.x >> 5) + (threadIdx.x >> 5);
  const int qt = wave & (CS / 16 - 1);        // 0..127
  const int h  = (wave >> 7) & (CH - 1);
  const int b  = wave >> 11;
  const int q0 = qt * 16;
  const int c  = lane & 15;
  const int ks = lane >> 4;

  const bf16_t* Qp = Qb + (size_t)(b * CH + h) * CS * CD;
  const bf16_t* Kp = Kb + (size_t)(b * CH + h) * CS * CD;
  const bf16_t* Vp = Vt + (size_t)(b * CH + h) * CD * CS;

  // Q as B-fragment of K*Q^T: col n = query c, K-dim = d (contiguous)
  v16bf qfrag[2];
#pragma unroll
  for (int kc = 0; kc < 2; ++kc)
    qfrag[kc] = load_frag(Qp + (size_t)(q0 + c) * CD + kc * 32 + ks * 8);

  const v8f vzero = {0.f,0.f,0.f,0.f,0.f,0.f,0.f,0.f};
  v8f ot[4];
#pragma unroll
  for (int i = 0; i < 4; ++i) ot[i] = vzero;
  float m_run = -3.0e38f, l_run = 0.f;
  const float scale = 0.125f;                  // 1/sqrt(64)
  const int qg = q0 + c;

  // stage K fragments for the first kv tile: ka[kt*2+kc]
  v16bf ka[4];
  {
    const int row1 = min(16 + c, CS - 1);
#pragma unroll
    for (int kc = 0; kc < 2; ++kc) {
      ka[kc]     = load_frag(Kp + (size_t)c * CD + kc * 32 + ks * 8);
      ka[2 + kc] = load_frag(Kp + (size_t)row1 * CD + kc * 32 + ks * 8);
    }
  }

  for (int jb = 0; jb < q0 + 16; jb += 32) {
    // ---- V fragments for this tile (consumed after softmax)
    const int col0 = jb + ks * 8;
    const int col1 = min(jb + 16 + ks * 8, CS - 8);
    v16bf va[4];
#pragma unroll
    for (int dc = 0; dc < 4; ++dc) {
      const bf16_t* vr = Vp + (size_t)(dc * 16 + c) * CS;
      va[dc] = cat8(*(const v8bf*)(vr + col0), *(const v8bf*)(vr + col1));
    }
    // ---- prefetch next tile's V columns
    const int pj = jb + 32;
    if (pj < q0 + 16) {
      __builtin_prefetch(Vp + (size_t)lane * CS + pj, 0, 3);
      __builtin_prefetch(Vp + (size_t)(lane + 32) * CS + pj, 0, 3);
    }
    // ---- S^T tiles from staged K: st0 = K[jb..+15]*Q^T, st1 = K[jb+16..+31]*Q^T
    v8f st0 = vzero, st1 = vzero;
#pragma unroll
    for (int kc = 0; kc < 2; ++kc) {
      st0 = WMMA_BF16(ka[kc],     qfrag[kc], st0);
      st1 = WMMA_BF16(ka[2 + kc], qfrag[kc], st1);
    }
    // ---- stage K fragments for the next tile (hidden behind softmax + PV)
    {
      const int njb  = (pj < q0 + 16) ? pj : jb;     // last iter: reload (discarded)
      const int nrow1 = min(njb + 16 + c, CS - 1);
#pragma unroll
      for (int kc = 0; kc < 2; ++kc) {
        ka[kc]     = load_frag(Kp + (size_t)(njb + c) * CD + kc * 32 + ks * 8);
        ka[2 + kc] = load_frag(Kp + (size_t)nrow1 * CD + kc * 32 + ks * 8);
      }
    }
    // ---- scale + causal mask (rows = kn, cols = q)
    float mx = -3.0e38f;
#pragma unroll
    for (int v = 0; v < 8; ++v) {
      const int kn = jb + v + 8 * ks;
      float s0 = (kn <= qg)      ? st0[v] * scale : -3.0e38f;
      float s1 = (kn + 16 <= qg) ? st1[v] * scale : -3.0e38f;
      st0[v] = s0; st1[v] = s1;
      mx = fmaxf(mx, fmaxf(s0, s1));
    }
    mx = fmaxf(mx, __shfl_xor(mx, 16));
    const float m_new = fmaxf(m_run, mx);
    const float alpha = __expf(m_run - m_new);
    // ---- P = exp(S^T - m); C-layout == B-fragment layout for the PV WMMA
    float rsum = 0.f;
    v16bf pb;
#pragma unroll
    for (int v = 0; v < 8; ++v) {
      float e0 = __expf(st0[v] - m_new);
      float e1 = __expf(st1[v] - m_new);
      rsum += e0 + e1;
      pb[v]     = (bf16_t)e0;
      pb[8 + v] = (bf16_t)e1;
    }
    rsum += __shfl_xor(rsum, 16);
    l_run = l_run * alpha + rsum;
    m_run = m_new;
    // ---- rescale O^T and accumulate O^T += V^T * P
#pragma unroll
    for (int dc = 0; dc < 4; ++dc) {
#pragma unroll
      for (int v = 0; v < 8; ++v) ot[dc][v] *= alpha;
    }
#pragma unroll
    for (int dc = 0; dc < 4; ++dc)
      ot[dc] = WMMA_BF16(va[dc], pb, ot[dc]);
  }

  const float inv = 1.f / l_run;
  // lane holds O^T[d = dc*16 + v + 8*ks][q = c]; pack 8 consecutive d -> b128
  bf16_t* orow = Ob + ((size_t)(b * CS + q0 + c)) * CE + h * CD;
#pragma unroll
  for (int dc = 0; dc < 4; ++dc) {
    v8bf o8;
#pragma unroll
    for (int v = 0; v < 8; ++v) o8[v] = (bf16_t)(ot[dc][v] * inv);
    *(v8bf*)(orow + dc * 16 + ks * 8) = o8;
  }
}

// ---------------------------------------------------------------------------
// Output projection: out[m,f] = sum_e O[m,e]*Wo[f,e] + b[f]  (f32 out)
// One wave -> 32(M) x 64(N), fully-unrolled ping-pong like gemm_qkv.
// ---------------------------------------------------------------------------
__global__ __launch_bounds__(256) void gemm_out_kernel(
    const bf16_t* __restrict__ Ob, const bf16_t* __restrict__ Wob,
    const float* __restrict__ bias, float* __restrict__ out) {
  const int lane = threadIdx.x & 31;
  const int wave = blockIdx.x * (blockDim.x >> 5) + (threadIdx.x >> 5);
  const int NT = CE / 64;                    // 16 n-tiles
  const int m0 = (wave / NT) * 32;
  const int n0 = (wave % NT) * 64;
  const int r  = lane & 15;
  const int ks = lane >> 4;

  v8f acc[8];
  const v8f vzero = {0.f,0.f,0.f,0.f,0.f,0.f,0.f,0.f};
#pragma unroll
  for (int i = 0; i < 8; ++i) acc[i] = vzero;

  const bf16_t* arow0 = Ob + (size_t)(m0 + r) * CE + ks * 8;
  const bf16_t* arow1 = Ob + (size_t)(m0 + 16 + r) * CE + ks * 8;
  const bf16_t* brow[4];
#pragma unroll
  for (int nn = 0; nn < 4; ++nn)
    brow[nn] = Wob + (size_t)(n0 + nn * 16 + r) * CE + ks * 8;

  v16bf pa0 = load_frag(arow0), pa1 = load_frag(arow1);
  v16bf pb[4];
#pragma unroll
  for (int nn = 0; nn < 4; ++nn) pb[nn] = load_frag(brow[nn]);

#pragma unroll
  for (int kk = 0; kk < CE; kk += 64) {
    const int k1 = kk + 32;
    v16bf qa0 = load_frag(arow0 + k1);
    v16bf qa1 = load_frag(arow1 + k1);
    v16bf qb[4];
#pragma unroll
    for (int nn = 0; nn < 4; ++nn) qb[nn] = load_frag(brow[nn] + k1);
#pragma unroll
    for (int nn = 0; nn < 4; ++nn) acc[nn]     = WMMA_BF16(pa0, pb[nn], acc[nn]);
#pragma unroll
    for (int nn = 0; nn < 4; ++nn) acc[4 + nn] = WMMA_BF16(pa1, pb[nn], acc[4 + nn]);
    if (kk + 64 < CE) {
      const int k2 = kk + 64;
      pa0 = load_frag(arow0 + k2);
      pa1 = load_frag(arow1 + k2);
#pragma unroll
      for (int nn = 0; nn < 4; ++nn) pb[nn] = load_frag(brow[nn] + k2);
    }
#pragma unroll
    for (int nn = 0; nn < 4; ++nn) acc[nn]     = WMMA_BF16(qa0, qb[nn], acc[nn]);
#pragma unroll
    for (int nn = 0; nn < 4; ++nn) acc[4 + nn] = WMMA_BF16(qa1, qb[nn], acc[4 + nn]);
  }

#pragma unroll
  for (int mt = 0; mt < 2; ++mt) {
#pragma unroll
    for (int nn = 0; nn < 4; ++nn) {
      const int f = n0 + nn * 16 + r;
      const float bv = bias[f];
#pragma unroll
      for (int v = 0; v < 8; ++v) {
        const int m = m0 + mt * 16 + v + 8 * ks;
        out[(size_t)m * CE + f] = acc[mt * 4 + nn][v] + bv;
      }
    }
  }
}

// ---------------------------------------------------------------------------
extern "C" void kernel_launch(void* const* d_in, const int* in_sizes, int n_in,
                              void* d_out, int out_size, void* d_ws, size_t ws_size,
                              hipStream_t stream) {
  const float* x    = (const float*)d_in[0];
  const float* Wa_w = (const float*)d_in[1];
  const float* Wa_b = (const float*)d_in[2];
  const float* Wo_w = (const float*)d_in[3];
  const float* Wo_b = (const float*)d_in[4];
  float* out = (float*)d_out;

  char* ws = (char*)d_ws;
  const size_t MB = 1024 * 1024;
  bf16_t* xb  = (bf16_t*)(ws);              // 16 MB : x bf16 [M,E]
  bf16_t* Wab = (bf16_t*)(ws + 16 * MB);    //  6 MB : Wa bf16 [3E,E]
  bf16_t* Wob = (bf16_t*)(ws + 22 * MB);    //  2 MB : Wo bf16 [E,E]
  bf16_t* Qb  = (bf16_t*)(ws + 24 * MB);    // 16 MB : Q [B,H,S,D]
  bf16_t* Kb  = (bf16_t*)(ws + 40 * MB);    // 16 MB : K [B,H,S,D]
  bf16_t* Vt  = (bf16_t*)(ws + 56 * MB);    // 16 MB : V^T [B,H,D,S]
  bf16_t* Ob  = (bf16_t*)(ws + 72 * MB);    // 16 MB : attn out [B,S,E]

  const int nx = CM * CE;            // 8388608
  const int nwa = 3 * CE * CE;       // 3145728
  const int nwo = CE * CE;           // 1048576
  cvt_bf16_kernel<<<(nx  + 255) / 256, 256, 0, stream>>>(x,    xb,  nx);
  cvt_bf16_kernel<<<(nwa + 255) / 256, 256, 0, stream>>>(Wa_w, Wab, nwa);
  cvt_bf16_kernel<<<(nwo + 255) / 256, 256, 0, stream>>>(Wo_w, Wob, nwo);

  // QKV GEMM: (8192/32) * (3072/64) = 12288 waves, 8 waves/block
  gemm_qkv_kernel<<<12288 / 8, 256, 0, stream>>>(xb, Wab, Wa_b, Qb, Kb, Vt);

  // Attention: B*H*(S/16) = 8192 waves
  attn_kernel<<<8192 / 8, 256, 0, stream>>>(Qb, Kb, Vt, Ob);

  // Output GEMM: (8192/32) * (1024/64) = 4096 waves
  gemm_out_kernel<<<4096 / 8, 256, 0, stream>>>(Ob, Wob, Wo_b, out);
}
